// NNConv_block_58291296141370
// MI455X (gfx1250) — matrix operand order repaired and checked by reading, not compiled, not run on hardware
//
#include <hip/hip_runtime.h>

// ---------------------------------------------------------------------------
// NNConv(edge-MLP) + scatter-mean + GRU for MI455X (gfx1250, wave32, WMMA).
//
// Fusion: never materialize per-edge 64x64 W (1.3GB HBM traffic avoided).
// msg = (u ⊗ x_src) @ Wl_flat, u = [BN(edge_attr), 1]:
//   one [E x 832] x [832 x 64] bf16 GEMM -> v_wmma_f32_16x16x32_bf16.
// Each wave: 32 edges (2 M-tiles) x 64 outputs, B fragments double-buffered
// so 8 WMMAs cover every global B load. GRU gate GEMMs also on WMMA.
// ---------------------------------------------------------------------------

#define N_NODES 20000
#define N_EDGES 80000
#define DIM 64
#define EDIM 12

typedef __attribute__((ext_vector_type(16))) __bf16 v16bf;
typedef __attribute__((ext_vector_type(8)))  __bf16 v8bf;
typedef __attribute__((ext_vector_type(2)))  __bf16 v2bf;
typedef __attribute__((ext_vector_type(8)))  float  v8f;

constexpr int KTOT = 13 * 64;   // 832 = fused K (12 BN features + bias row) * 64
constexpr int NKB  = KTOT / 32; // 26 k-blocks of 32 (even -> clean ping-pong)

// Workspace layout (bytes). bf16 matrices 32B aligned for v16bf loads.
constexpr size_t OFF_BNACC = 0;                                  // 32 f32
constexpr size_t OFF_SS    = 128;                                // 32 f32
constexpr size_t OFF_CNT   = 256;                                // N f32
constexpr size_t OFF_AGG   = OFF_CNT + (size_t)N_NODES * 4;      // 80256
constexpr size_t OFF_BT    = OFF_AGG + (size_t)N_NODES * DIM * 4;// 5200256
constexpr size_t OFF_WIH   = OFF_BT  + (size_t)DIM * KTOT * 2;   // 5306752
constexpr size_t OFF_WHH   = OFF_WIH + (size_t)3 * DIM * DIM * 2;// 5331328
constexpr size_t OFF_XBF   = OFF_WHH + (size_t)3 * DIM * DIM * 2;// 5355904
// total ~7.92 MB of d_ws

__global__ void k_zero(float* p, int n) {
  int i = blockIdx.x * blockDim.x + threadIdx.x;
  if (i < n) p[i] = 0.0f;
}

__global__ __launch_bounds__(256) void k_bnstats(const float* __restrict__ ea,
                                                 float* __restrict__ acc) {
  __shared__ float red[256];
  float s[EDIM], q[EDIM];
#pragma unroll
  for (int k = 0; k < EDIM; ++k) { s[k] = 0.f; q[k] = 0.f; }
  for (int e = blockIdx.x * blockDim.x + threadIdx.x; e < N_EDGES;
       e += gridDim.x * blockDim.x) {
    const float* row = ea + (size_t)e * EDIM;
#pragma unroll
    for (int k = 0; k < EDIM; ++k) { float v = row[k]; s[k] += v; q[k] += v * v; }
  }
#pragma unroll
  for (int c = 0; c < 2 * EDIM; ++c) {
    red[threadIdx.x] = (c < EDIM) ? s[c] : q[c - EDIM];
    __syncthreads();
    for (int off = 128; off > 0; off >>= 1) {
      if ((int)threadIdx.x < off) red[threadIdx.x] += red[threadIdx.x + off];
      __syncthreads();
    }
    if (threadIdx.x == 0) atomicAdd(acc + c, red[0]);
    __syncthreads();
  }
}

__global__ void k_bnfin(const float* acc, const float* gamma, const float* beta,
                        float* ss) {
  int k = threadIdx.x;
  if (k < EDIM) {
    float mu  = acc[k] * (1.0f / N_EDGES);
    float var = acc[EDIM + k] * (1.0f / N_EDGES) - mu * mu;
    float sc  = gamma[k] * rsqrtf(var + 1e-5f);
    ss[k]      = sc;
    ss[16 + k] = beta[k] - mu * sc;
  }
}

// Bt[o][K] transposed so each lane's B fragment is a contiguous 32B K-run.
__global__ void k_build_bt(const float* __restrict__ lw, const float* __restrict__ lb,
                           __bf16* __restrict__ bt) {
  int idx = blockIdx.x * blockDim.x + threadIdx.x;
  if (idx >= DIM * KTOT) return;
  int o = idx / KTOT, K = idx % KTOT;
  int k13 = K >> 6, i = K & 63;
  float v = (k13 < EDIM) ? lw[(size_t)k13 * (DIM * DIM) + i * DIM + o]
                         : lb[i * DIM + o];
  bt[idx] = (__bf16)v;
}

__global__ void k_cvt_bf16(const float* __restrict__ src, __bf16* __restrict__ dst, int n) {
  int i = blockIdx.x * blockDim.x + threadIdx.x;
  if (i < n) dst[i] = (__bf16)src[i];
}

// ---------------------------------------------------------------------------
// Fused message kernel: 1 wave per 32-edge tile (2 M-tiles), 4 waves/block.
// A built on the fly as u[k13]*x_src[i]; B double-buffered from L2-resident Bt.
// 26 k-blocks x 4 n-tiles x 2 m-tiles = 208 WMMAs per wave.
// ---------------------------------------------------------------------------
__global__ __launch_bounds__(128) void k_msg(
    const int* __restrict__ ei, const float* __restrict__ ea,
    const float* __restrict__ ss, const __bf16* __restrict__ bt,
    const __bf16* __restrict__ xbf, float* __restrict__ agg,
    float* __restrict__ cnt) {
  __shared__ __align__(16) __bf16 shx[4][32][DIM];  // staged x_src rows
  __shared__ float shu[4][32][16];                  // BN'd edge feats + bias 1
  __shared__ int   shd[4][32];                      // dst indices
  const int wave = threadIdx.x >> 5, lane = threadIdx.x & 31;
  const int e0 = (blockIdx.x * 4 + wave) * 32;      // grid exact: E/32 tiles

  {  // stage: every lane owns one edge
    int e = e0 + lane;
    shd[wave][lane] = ei[N_EDGES + e];              // row 1 = dst
#pragma unroll
    for (int k = 0; k < EDIM; ++k)
      shu[wave][lane][k] = ea[(size_t)e * EDIM + k] * ss[k] + ss[16 + k];
    shu[wave][lane][EDIM] = 1.0f;                   // bias row of fused K
    int srow = ei[e];                               // row 0 = src
    const uint4* xr4 = (const uint4*)(xbf + (size_t)srow * DIM);
    uint4* d4 = (uint4*)&shx[wave][lane][0];
#pragma unroll
    for (int q = 0; q < 4; ++q) d4[q] = xr4[q];     // 64B half-row copy, b128 ops
  }
  __syncthreads();

  const int m    = lane & 15;          // A row within M-tile
  const int ahi  = (lane >> 4) << 3;   // +8  M offset, upper half-wave (C layout)
  const int bhi  = (lane >> 4) << 4;   // +16 K offset, upper half-wave (B layout)
  const int ncol = lane & 15;          // B/C column

  v8f c[8];
#pragma unroll
  for (int i = 0; i < 8; ++i) c[i] = {};

  // B fragment loader: 2 contiguous b128 per n-tile
  auto loadB = [&](int kb, v16bf* dst) {
    const int krow = kb * 32 + bhi;
#pragma unroll
    for (int nt = 0; nt < 4; ++nt)
      dst[nt] = *(const v16bf*)(bt + (size_t)(nt * 16 + ncol) * KTOT + krow);
  };
  // A fragment: two contiguous 16B LDS runs, scaled by u and cvt to bf16
  auto buildA = [&](int rowbase, int k13, int ibase) -> v16bf {
    const float uv = shu[wave][rowbase + m][k13];
    v8bf lo = *(const v8bf*)&shx[wave][rowbase + m][ibase + bhi];
    v8bf hi = *(const v8bf*)&shx[wave][rowbase + m][ibase + 8 + bhi];
    v16bf a;
#pragma unroll
    for (int t = 0; t < 8; ++t) {
      a[t]     = (__bf16)(uv * (float)lo[t]);
      a[t + 8] = (__bf16)(uv * (float)hi[t]);
    }
    return a;
  };
  auto doKB = [&](int kb, const v16bf* b) {
    const int k13 = kb >> 1, ibase = (kb & 1) * 32;
    v16bf a0 = buildA(0, k13, ibase);
    v16bf a1 = buildA(16, k13, ibase);
#pragma unroll
    for (int nt = 0; nt < 4; ++nt)
      c[nt] = __builtin_amdgcn_wmma_f32_16x16x32_bf16(
          false, a0, false, b[nt], (short)0, c[nt], false, false);
#pragma unroll
    for (int nt = 0; nt < 4; ++nt)
      c[4 + nt] = __builtin_amdgcn_wmma_f32_16x16x32_bf16(
          false, a1, false, b[nt], (short)0, c[4 + nt], false, false);
  };

  v16bf bA[4], bB[4];
  loadB(0, bA);
  for (int kb = 0; kb < NKB; kb += 2) {   // ping-pong: loads lead WMMAs by 8
    loadB(kb + 1, bB);
    doKB(kb, bA);
    if (kb + 2 < NKB) loadB(kb + 2, bA);
    doKB(kb + 1, bB);
  }

  // scatter-mean numerator (C layout: M = v + 8*(lane>>4)) + denominator
#pragma unroll
  for (int nt = 0; nt < 4; ++nt)
#pragma unroll
    for (int v = 0; v < 8; ++v) {
      int r0 = v + ahi;
      atomicAdd(agg + (size_t)shd[wave][r0]      * DIM + nt * 16 + ncol, c[nt][v]);
      atomicAdd(agg + (size_t)shd[wave][16 + r0] * DIM + nt * 16 + ncol, c[4 + nt][v]);
    }
  atomicAdd(cnt + shd[wave][lane], 1.0f);
}

// ---------------------------------------------------------------------------
// Fused mean + ReLU + GRU step: 1 wave per 16 nodes.
// gi = m @ w_ih^T + b_ih (WMMA -> LDS), gh = x @ w_hh^T (WMMA, regs),
// fragment-wise gate math, f32 store.
// ---------------------------------------------------------------------------
__global__ __launch_bounds__(128) void k_gru(
    const float* __restrict__ x, const __bf16* __restrict__ xbf,
    const float* __restrict__ agg, const float* __restrict__ cnt,
    const __bf16* __restrict__ wih, const __bf16* __restrict__ whh,
    const float* __restrict__ bih, const float* __restrict__ bhh,
    float* __restrict__ out) {
  __shared__ float shgi[4][16][3 * DIM];
  const int wave = threadIdx.x >> 5, lane = threadIdx.x & 31;
  const int node0 = (blockIdx.x * 4 + wave) * 16;
  const int m    = lane & 15;
  const int ahi  = (lane >> 4) << 3;
  const int bhi  = (lane >> 4) << 4;
  const int ncol = lane & 15;
  int node_m = node0 + m; if (node_m >= N_NODES) node_m = N_NODES - 1; // clamp tail
  const float inv = 1.0f / fmaxf(cnt[node_m], 1.0f);

  // Both A fragments (K=64 -> 2 k-blocks): am = relu(agg/cnt), ax = x.
  v16bf am[2], ax[2];
#pragma unroll
  for (int kb = 0; kb < 2; ++kb) {
#pragma unroll
    for (int t = 0; t < 16; t += 2) {
      int i = kb * 32 + t + bhi;
      am[kb][t]     = (__bf16)fmaxf(agg[(size_t)node_m * DIM + i]     * inv, 0.0f);
      am[kb][t + 1] = (__bf16)fmaxf(agg[(size_t)node_m * DIM + i + 1] * inv, 0.0f);
      v2bf p = *(const v2bf*)(xbf + (size_t)node_m * DIM + i);
      ax[kb][t] = p[0]; ax[kb][t + 1] = p[1];
    }
  }

  v8f acc[12];
#pragma unroll
  for (int nt = 0; nt < 12; ++nt) acc[nt] = {};
#pragma unroll
  for (int kb = 0; kb < 2; ++kb) {
    const int krow = kb * 32 + bhi;
#pragma unroll
    for (int nt = 0; nt < 12; ++nt) {
      v16bf b = *(const v16bf*)(wih + (size_t)(nt * 16 + ncol) * DIM + krow);
      acc[nt] = __builtin_amdgcn_wmma_f32_16x16x32_bf16(
          false, am[kb], false, b, (short)0, acc[nt], false, false);
    }
  }
#pragma unroll
  for (int nt = 0; nt < 12; ++nt) {
    float bi = bih[nt * 16 + ncol];
#pragma unroll
    for (int v = 0; v < 8; ++v)
      shgi[wave][v + ahi][nt * 16 + ncol] = acc[nt][v] + bi;
  }
  __syncthreads();

#pragma unroll
  for (int nt = 0; nt < 12; ++nt) acc[nt] = {};
#pragma unroll
  for (int kb = 0; kb < 2; ++kb) {
    const int krow = kb * 32 + bhi;
#pragma unroll
    for (int nt = 0; nt < 12; ++nt) {
      v16bf b = *(const v16bf*)(whh + (size_t)(nt * 16 + ncol) * DIM + krow);
      acc[nt] = __builtin_amdgcn_wmma_f32_16x16x32_bf16(
          false, ax[kb], false, b, (short)0, acc[nt], false, false);
    }
  }

  // Gate math: torch order r,z,n.  d = nt*16 + ncol.
#pragma unroll
  for (int nt = 0; nt < 4; ++nt) {
    const int g = nt * 16 + ncol;
    const float bhr = bhh[g], bhz = bhh[64 + g], bhn = bhh[128 + g];
#pragma unroll
    for (int v = 0; v < 8; ++v) {
      const int row  = v + ahi;
      const int node = node0 + row;
      float ir  = shgi[wave][row][g];
      float iz  = shgi[wave][row][64 + g];
      float inn = shgi[wave][row][128 + g];
      float r  = 1.0f / (1.0f + __expf(-(ir + acc[nt][v] + bhr)));
      float zz = 1.0f / (1.0f + __expf(-(iz + acc[nt + 4][v] + bhz)));
      float nh = tanhf(inn + r * (acc[nt + 8][v] + bhn));
      if (node < N_NODES) {
        float xv = x[(size_t)node * DIM + g];
        out[(size_t)node * DIM + g] = (1.0f - zz) * nh + zz * xv;
      }
    }
  }
}

extern "C" void kernel_launch(void* const* d_in, const int* in_sizes, int n_in,
                              void* d_out, int out_size, void* d_ws, size_t ws_size,
                              hipStream_t stream) {
  const float* x     = (const float*)d_in[0];
  const int*   ei    = (const int*)  d_in[1];
  const float* ea    = (const float*)d_in[2];
  const float* gamma = (const float*)d_in[3];
  const float* beta  = (const float*)d_in[4];
  const float* lin_w = (const float*)d_in[5];
  const float* lin_b = (const float*)d_in[6];
  const float* w_ih  = (const float*)d_in[7];
  const float* w_hh  = (const float*)d_in[8];
  const float* b_ih  = (const float*)d_in[9];
  const float* b_hh  = (const float*)d_in[10];
  float* out = (float*)d_out;

  char* ws = (char*)d_ws;
  float*  bnacc = (float*)(ws + OFF_BNACC);
  float*  ssp   = (float*)(ws + OFF_SS);
  float*  cntp  = (float*)(ws + OFF_CNT);
  float*  aggp  = (float*)(ws + OFF_AGG);
  __bf16* btp   = (__bf16*)(ws + OFF_BT);
  __bf16* wihp  = (__bf16*)(ws + OFF_WIH);
  __bf16* whhp  = (__bf16*)(ws + OFF_WHH);
  __bf16* xbfp  = (__bf16*)(ws + OFF_XBF);

  const int nzero = (int)(OFF_BT / 4);   // zero [bnacc|ss|cnt|agg]
  k_zero<<<(nzero + 255) / 256, 256, 0, stream>>>((float*)ws, nzero);

  k_bnstats<<<128, 256, 0, stream>>>(ea, bnacc);
  k_bnfin<<<1, 32, 0, stream>>>(bnacc, gamma, beta, ssp);

  k_build_bt<<<(DIM * KTOT + 255) / 256, 256, 0, stream>>>(lin_w, lin_b, btp);
  k_cvt_bf16<<<(3 * DIM * DIM + 255) / 256, 256, 0, stream>>>(w_ih, wihp, 3 * DIM * DIM);
  k_cvt_bf16<<<(3 * DIM * DIM + 255) / 256, 256, 0, stream>>>(w_hh, whhp, 3 * DIM * DIM);
  k_cvt_bf16<<<(N_NODES * DIM + 255) / 256, 256, 0, stream>>>(x, xbfp, N_NODES * DIM);

  k_msg<<<N_EDGES / 128, 128, 0, stream>>>(ei, ea, ssp, btp, xbfp, aggp, cntp);
  k_gru<<<(N_NODES + 63) / 64, 128, 0, stream>>>(x, xbfp, aggp, cntp, wihp, whhp,
                                                 b_ih, b_hh, out);
}
